// MultiheadAttention_15487652070057
// MI455X (gfx1250) — matrix operand order repaired and checked by reading, not compiled, not run on hardware
//
#include <hip/hip_runtime.h>
#include <hip/hip_bf16.h>
#include <math.h>

// ---------------------------------------------------------------------------
// MultiheadAttention for MI455X (gfx1250): bf16 WMMA everywhere.
// B=2, S=2048, D_IN=D=1024, H=16, HD=64.
// Round 2: software-pipelined (double-buffered) GEMM k-loops; attention
// issues V-frag loads before the softmax VALU pass so global
// latency overlaps with exp/reductions.
// ---------------------------------------------------------------------------

typedef __attribute__((ext_vector_type(16))) __bf16 v16bf;
typedef __attribute__((ext_vector_type(8)))  __bf16 v8bf;
typedef __attribute__((ext_vector_type(8)))  float  v8f;

#define S_LEN   2048
#define D_MODEL 1024
#define NHEAD   16
#define HDIM    64
#define N_QKV   3072
#define M_ROWS  4096   // B * S

__device__ __forceinline__ v8f vzero8() {
  v8f z;
#pragma unroll
  for (int i = 0; i < 8; ++i) z[i] = 0.0f;
  return z;
}

__device__ __forceinline__ v8f wmma_bf16(v16bf a, v16bf b, v8f c) {
  // D = A(16x32) * B(32x16) + C, f32 accumulate
  return __builtin_amdgcn_wmma_f32_16x16x32_bf16(
      /*neg_a=*/false, a, /*neg_b=*/false, b,
      /*c_mod=*/(short)0, c, /*reuse_a=*/false, /*reuse_b=*/false);
}

// A-matrix fragment: 16x32 bf16 tile from row-major [rows, ld] source.
// ISA layout: lanes 0-15 row M=lane, VGPR0-3 = K 0..7, VGPR4-7 = K 16..23;
//             lanes 16-31 same rows, K 8..15 and 24..31.
__device__ __forceinline__ v16bf ldfragA(const __bf16* p, int ld, int lane) {
  const int lm = lane & 15, hi = lane >> 4;
  const __bf16* q = p + (size_t)lm * ld + hi * 8;
  v8bf c0 = *(const v8bf*)(q);        // K = hi*8 + 0..7
  v8bf c1 = *(const v8bf*)(q + 16);   // K = 16 + hi*8 + 0..7
  v16bf r;
#pragma unroll
  for (int i = 0; i < 8; ++i) { r[i] = c0[i]; r[i + 8] = c1[i]; }
  return r;
}

// B-matrix fragment: 32x16 bf16 tile (K x N), sourced NT-style from a
// row-major [N, ld] matrix (column n of B == row n of W).
// ISA layout (matches A): lanes 0-15 (col N=lane) hold K 0..7 and 16..23;
// lanes 16-31 hold K 8..15 and 24..31.
__device__ __forceinline__ v16bf ldfragB(const __bf16* p, int ld, int lane) {
  const int n = lane & 15, half = lane >> 4;
  const __bf16* q = p + (size_t)n * ld + half * 8;
  v8bf c0 = *(const v8bf*)(q);        // K = half*8 + 0..7
  v8bf c1 = *(const v8bf*)(q + 16);   // K = 16 + half*8 + 0..7
  v16bf r;
#pragma unroll
  for (int i = 0; i < 8; ++i) { r[i] = c0[i]; r[i + 8] = c1[i]; }
  return r;
}

// Shared GEMM helpers: wave computes a 64x32 tile as 4x2 WMMA tiles, both
// operands row-major with leading dim D_MODEL (NT GEMM).
__device__ __forceinline__ void load_ab(const __bf16* __restrict__ am,
                                        const __bf16* __restrict__ bm,
                                        int mBase, int nBase, int kc, int lane,
                                        v16bf a[4], v16bf b[2]) {
#pragma unroll
  for (int i = 0; i < 4; ++i)
    a[i] = ldfragA(am + (size_t)(mBase + 16 * i) * D_MODEL + kc, D_MODEL, lane);
#pragma unroll
  for (int j = 0; j < 2; ++j)
    b[j] = ldfragB(bm + (size_t)(nBase + 16 * j) * D_MODEL + kc, D_MODEL, lane);
}

__device__ __forceinline__ void mma_tiles(const v16bf a[4], const v16bf b[2],
                                          v8f acc[4][2]) {
#pragma unroll
  for (int i = 0; i < 4; ++i)
#pragma unroll
    for (int j = 0; j < 2; ++j)
      acc[i][j] = wmma_bf16(a[i], b[j], acc[i][j]);
}

// ---------------------------------------------------------------------------
// f32 -> bf16 conversion
// ---------------------------------------------------------------------------
__global__ void cvt_f32_bf16(const float* __restrict__ src,
                             __bf16* __restrict__ dst, int n) {
  int i = blockIdx.x * blockDim.x + threadIdx.x;
  if (i < n) dst[i] = (__bf16)src[i];
}

// ---------------------------------------------------------------------------
// QKV projection: [4096,1024] x [3072,1024]^T -> routed into
//   Qs [B,H,S,HD] (pre-scaled by 1/sqrt(HD)), Ks [B,H,S,HD], Vt [B,H,HD,S]
// Block: 128x128 tile; 8 waves (2 M x 4 N); double-buffered k-loop.
// ---------------------------------------------------------------------------
__global__ __launch_bounds__(256) void qkv_gemm(
    const __bf16* __restrict__ xb, const __bf16* __restrict__ wqb,
    const float* __restrict__ bqkv,
    __bf16* __restrict__ Qs, __bf16* __restrict__ Ks, __bf16* __restrict__ Vt) {
  const int lane = threadIdx.x & 31;
  const int wave = threadIdx.x >> 5;
  const int mBase = blockIdx.y * 128 + (wave >> 2) * 64;
  const int nBase = blockIdx.x * 128 + (wave & 3) * 32;

  v8f acc[4][2];
#pragma unroll
  for (int i = 0; i < 4; ++i)
#pragma unroll
    for (int j = 0; j < 2; ++j) acc[i][j] = vzero8();

  v16bf a0[4], b0[2], a1[4], b1[2];
  load_ab(xb, wqb, mBase, nBase, 0, lane, a0, b0);
  for (int kc = 0; kc < D_MODEL; kc += 64) {
    // speculative prefetch two k-steps ahead (safe: dropped on xlat failure)
    __builtin_prefetch(xb + (size_t)mBase * D_MODEL + kc + 128, 0, 1);
    __builtin_prefetch(wqb + (size_t)nBase * D_MODEL + kc + 128, 0, 1);
    load_ab(xb, wqb, mBase, nBase, kc + 32, lane, a1, b1);   // next half
    mma_tiles(a0, b0, acc);
    if (kc + 64 < D_MODEL)
      load_ab(xb, wqb, mBase, nBase, kc + 64, lane, a0, b0); // next iter
    mma_tiles(a1, b1, acc);
  }

  const int ln = lane & 15, hi = lane >> 4;
#pragma unroll
  for (int i = 0; i < 4; ++i) {
#pragma unroll
    for (int j = 0; j < 2; ++j) {
      const int n = nBase + 16 * j + ln;
      const int h = n / 192, c = n - h * 192;
      const float bias = bqkv[n];
#pragma unroll
      for (int r = 0; r < 8; ++r) {
        const int m = mBase + 16 * i + r + 8 * hi;
        const int bidx = m >> 11;           // / S_LEN
        const int s = m & (S_LEN - 1);
        const size_t bh = (size_t)bidx * NHEAD + h;
        const float v = acc[i][j][r] + bias;
        if (c < HDIM) {
          Qs[(bh * S_LEN + s) * HDIM + c] = (__bf16)(v * 0.125f); // 1/sqrt(64)
        } else if (c < 2 * HDIM) {
          Ks[(bh * S_LEN + s) * HDIM + (c - HDIM)] = (__bf16)v;
        } else {
          Vt[(bh * HDIM + (c - 2 * HDIM)) * S_LEN + s] = (__bf16)v; // transposed
        }
      }
    }
  }
}

// ---------------------------------------------------------------------------
// Flash attention: one wave per 16-query tile. Online softmax (base-2) in f32.
// Per 32-key block: 4 WMMA for Q*K^T, LDS round-trip for P layout, 4 WMMA for
// P*V. K-fragments are pipelined one block ahead; V-fragments are issued
// before the softmax VALU pass so their latency hides under exp/reductions.
// ---------------------------------------------------------------------------
__global__ __launch_bounds__(128) void attn_kernel(
    const __bf16* __restrict__ Qs, const __bf16* __restrict__ Ks,
    const __bf16* __restrict__ Vt, __bf16* __restrict__ vals) {
  __shared__ __align__(16) __bf16 pbuf[4][16][32];   // per-wave P tile

  const int lane = threadIdx.x & 31;
  const int wave = threadIdx.x >> 5;
  const int ln = lane & 15, hi = lane >> 4;
  const int tile = blockIdx.x * 4 + wave;            // 0..4095
  const int bh = tile >> 7;                          // b*H + h
  const int qt = tile & 127;
  const int bidx = bh >> 4;
  const int h = bh & 15;

  const __bf16* Qbase = Qs + ((size_t)bh * S_LEN + (size_t)qt * 16) * HDIM;
  const __bf16* Kbh = Ks + (size_t)bh * S_LEN * HDIM;
  const __bf16* Vbh = Vt + (size_t)bh * HDIM * S_LEN;

  const v16bf q0 = ldfragA(Qbase, HDIM, lane);        // K dim 0..31
  const v16bf q1 = ldfragA(Qbase + 32, HDIM, lane);   // K dim 32..63

  v8f oacc[4];
#pragma unroll
  for (int t = 0; t < 4; ++t) oacc[t] = vzero8();
  float rowm[8], rowsum[8];
#pragma unroll
  for (int r = 0; r < 8; ++r) { rowm[r] = -3.0e38f; rowsum[r] = 0.0f; }

  const float L2E = 1.4426950408889634f;

  // pipelined K fragments for the first key block
  v16bf kf[4];
  kf[0] = ldfragB(Kbh, HDIM, lane);
  kf[1] = ldfragB(Kbh + 32, HDIM, lane);
  kf[2] = ldfragB(Kbh + (size_t)16 * HDIM, HDIM, lane);
  kf[3] = ldfragB(Kbh + (size_t)16 * HDIM + 32, HDIM, lane);

  for (int kb = 0; kb < S_LEN; kb += 32) {
    // logits for keys kb..kb+15 (s0) and kb+16..kb+31 (s1)
    v8f s0 = vzero8(), s1 = vzero8();
    s0 = wmma_bf16(q0, kf[0], s0);
    s0 = wmma_bf16(q1, kf[1], s0);
    s1 = wmma_bf16(q0, kf[2], s1);
    s1 = wmma_bf16(q1, kf[3], s1);

    // issue V-fragment loads now; consumed only after softmax + LDS bounce
    v16bf vb[4];
#pragma unroll
    for (int t4 = 0; t4 < 4; ++t4)
      vb[t4] = ldfragB(Vbh + (size_t)(t4 * 16) * S_LEN + kb, S_LEN, lane);

#pragma unroll
    for (int r = 0; r < 8; ++r) {
      // row max over 32 keys: pairwise then across 16 lanes of the half-wave
      float t = fmaxf(s0[r], s1[r]);
      t = fmaxf(t, __shfl_xor(t, 1, 32));
      t = fmaxf(t, __shfl_xor(t, 2, 32));
      t = fmaxf(t, __shfl_xor(t, 4, 32));
      t = fmaxf(t, __shfl_xor(t, 8, 32));
      const float mn = fmaxf(rowm[r], t);
      const float corr = exp2f(L2E * (rowm[r] - mn));
      rowm[r] = mn;
      const float p0 = exp2f(L2E * (s0[r] - mn));
      const float p1 = exp2f(L2E * (s1[r] - mn));
      float ps = p0 + p1;
      ps += __shfl_xor(ps, 1, 32);
      ps += __shfl_xor(ps, 2, 32);
      ps += __shfl_xor(ps, 4, 32);
      ps += __shfl_xor(ps, 8, 32);
      rowsum[r] = rowsum[r] * corr + ps;
#pragma unroll
      for (int t4 = 0; t4 < 4; ++t4) oacc[t4][r] *= corr;
      // stash P (row m = r+8*hi, col = key-in-block) for A-fragment reload
      pbuf[wave][r + 8 * hi][ln] = (__bf16)p0;
      pbuf[wave][r + 8 * hi][16 + ln] = (__bf16)p1;
    }
    __syncthreads();   // uniform trip count across all 4 waves

    const v16bf pf = ldfragA(&pbuf[wave][0][0], 32, lane);  // P as 16x32 A-frag

    // prefetch next key block's K fragments before the PV WMMAs
    const int kn = kb + 32;
    if (kn < S_LEN) {
      kf[0] = ldfragB(Kbh + (size_t)kn * HDIM, HDIM, lane);
      kf[1] = ldfragB(Kbh + (size_t)kn * HDIM + 32, HDIM, lane);
      kf[2] = ldfragB(Kbh + (size_t)(kn + 16) * HDIM, HDIM, lane);
      kf[3] = ldfragB(Kbh + (size_t)(kn + 16) * HDIM + 32, HDIM, lane);
    }

#pragma unroll
    for (int t4 = 0; t4 < 4; ++t4)
      oacc[t4] = wmma_bf16(pf, vb[t4], oacc[t4]);
  }

  const size_t orow0 = (size_t)bidx * S_LEN + (size_t)qt * 16;
#pragma unroll
  for (int r = 0; r < 8; ++r) {
    const float inv = 1.0f / rowsum[r];
    const int m = r + 8 * hi;
#pragma unroll
    for (int t4 = 0; t4 < 4; ++t4) {
      vals[(orow0 + m) * D_MODEL + h * HDIM + t4 * 16 + ln] =
          (__bf16)(oacc[t4][r] * inv);
    }
  }
}

// ---------------------------------------------------------------------------
// Output projection: [4096,1024] x [1024,1024]^T + b_o -> f32 out
// ---------------------------------------------------------------------------
__global__ __launch_bounds__(256) void oproj_gemm(
    const __bf16* __restrict__ vals, const __bf16* __restrict__ wob,
    const float* __restrict__ bo, float* __restrict__ out) {
  const int lane = threadIdx.x & 31;
  const int wave = threadIdx.x >> 5;
  const int mBase = blockIdx.y * 128 + (wave >> 2) * 64;
  const int nBase = blockIdx.x * 128 + (wave & 3) * 32;

  v8f acc[4][2];
#pragma unroll
  for (int i = 0; i < 4; ++i)
#pragma unroll
    for (int j = 0; j < 2; ++j) acc[i][j] = vzero8();

  v16bf a0[4], b0[2], a1[4], b1[2];
  load_ab(vals, wob, mBase, nBase, 0, lane, a0, b0);
  for (int kc = 0; kc < D_MODEL; kc += 64) {
    __builtin_prefetch(vals + (size_t)mBase * D_MODEL + kc + 128, 0, 1);
    __builtin_prefetch(wob + (size_t)nBase * D_MODEL + kc + 128, 0, 1);
    load_ab(vals, wob, mBase, nBase, kc + 32, lane, a1, b1);
    mma_tiles(a0, b0, acc);
    if (kc + 64 < D_MODEL)
      load_ab(vals, wob, mBase, nBase, kc + 64, lane, a0, b0);
    mma_tiles(a1, b1, acc);
  }

  const int ln = lane & 15, hi = lane >> 4;
#pragma unroll
  for (int i = 0; i < 4; ++i)
#pragma unroll
    for (int j = 0; j < 2; ++j) {
      const int n = nBase + 16 * j + ln;
      const float bias = bo[n];
#pragma unroll
      for (int r = 0; r < 8; ++r) {
        const size_t m = (size_t)(mBase + 16 * i + r + 8 * hi);
        out[m * D_MODEL + n] = acc[i][j][r] + bias;
      }
    }
}

// ---------------------------------------------------------------------------
// Workspace layout (bytes):
//   xb   @ 0         (8,388,608)   x as bf16            [4096,1024]
//   wqb  @ 8388608   (6,291,456)   w_qkv as bf16        [3072,1024]
//   wob  @ 14680064  (2,097,152)   w_o as bf16          [1024,1024]
//   Qs   @ 16777216  (8,388,608)   Q scaled, bf16       [B,H,S,HD]
//   Ks   @ 25165824  (8,388,608)   K bf16               [B,H,S,HD]
//   Vt   @ 33554432  (8,388,608)   V transposed, bf16   [B,H,HD,S]
//   vals @ 41943040  (8,388,608)   attention out, bf16  [4096,1024]
// Total 48 MB.
// ---------------------------------------------------------------------------
extern "C" void kernel_launch(void* const* d_in, const int* in_sizes, int n_in,
                              void* d_out, int out_size, void* d_ws, size_t ws_size,
                              hipStream_t stream) {
  (void)in_sizes; (void)n_in; (void)out_size; (void)ws_size;
  const float* x     = (const float*)d_in[0];
  const float* w_qkv = (const float*)d_in[1];
  const float* b_qkv = (const float*)d_in[2];
  const float* w_o   = (const float*)d_in[3];
  const float* b_o   = (const float*)d_in[4];

  char* ws = (char*)d_ws;
  __bf16* xb   = (__bf16*)(ws + 0);
  __bf16* wqb  = (__bf16*)(ws + 8388608);
  __bf16* wob  = (__bf16*)(ws + 14680064);
  __bf16* Qs   = (__bf16*)(ws + 16777216);
  __bf16* Ks   = (__bf16*)(ws + 25165824);
  __bf16* Vt   = (__bf16*)(ws + 33554432);
  __bf16* vals = (__bf16*)(ws + 41943040);

  const int nx  = M_ROWS * D_MODEL;   // 4,194,304
  const int nwq = N_QKV * D_MODEL;    // 3,145,728
  const int nwo = D_MODEL * D_MODEL;  // 1,048,576
  cvt_f32_bf16<<<nx / 256, 256, 0, stream>>>(x, xb, nx);
  cvt_f32_bf16<<<nwq / 256, 256, 0, stream>>>(w_qkv, wqb, nwq);
  cvt_f32_bf16<<<nwo / 256, 256, 0, stream>>>(w_o, wob, nwo);

  qkv_gemm<<<dim3(N_QKV / 128, M_ROWS / 128), 256, 0, stream>>>(
      xb, wqb, b_qkv, Qs, Ks, Vt);

  attn_kernel<<<(2 * NHEAD * (S_LEN / 16)) / 4, 128, 0, stream>>>(
      Qs, Ks, Vt, vals);

  oproj_gemm<<<dim3(D_MODEL / 128, M_ROWS / 128), 256, 0, stream>>>(
      vals, wob, b_o, (float*)d_out);
}
